// KnotAttention_6141803233388
// MI455X (gfx1250) — matrix operand order repaired and checked by reading, not compiled, not run on hardware
//
#include <hip/hip_runtime.h>

#define WAVES_PER_BLOCK 8
#define NB_SOFT 240

#if __has_builtin(__builtin_amdgcn_global_load_async_to_lds_b128) && \
    __has_builtin(__builtin_amdgcn_s_wait_asynccnt)
#define USE_ASYNC_LDS 1
#else
#define USE_ASYNC_LDS 0
#endif

typedef __attribute__((ext_vector_type(16))) __bf16 v16bf;
typedef __attribute__((ext_vector_type(8)))  float  v8f;
typedef __attribute__((ext_vector_type(4)))  int    v4i;

__device__ __forceinline__ v8f vzero8() {
    v8f z;
#pragma unroll
    for (int i = 0; i < 8; ++i) z[i] = 0.0f;
    return z;
}

// A-fragment: 16x32 bf16, M = lane%16 (row pointer passed per lane), from global.
// lanes 0-15: K = kc*32 + [0..7] and [16..23]; lanes 16-31: +8.
__device__ __forceinline__ v16bf load_frag_a(const __bf16* rowp, int lane, int kc) {
    const char* p = (const char*)rowp + (kc << 6) + ((lane >> 4) << 4);
    v16bf a;
    ((uint4*)&a)[0] = *(const uint4*)(p);
    ((uint4*)&a)[1] = *(const uint4*)(p + 32);
    return a;
}

// B-fragment: 32x16 bf16 from LDS-staged W^T (row = output col, 128 K's per row).
// Rows stored at row*256 bytes; 16B chunks XOR-swizzled by (row&15) for bank-conflict-free ds loads.
// lanes 0-15: col = nt*16+lane, K = kc*32 + [0..7] and [16..23]; lanes 16-31: +8.
__device__ __forceinline__ v16bf load_frag_b_lds(const char* base, int rowbase, int nt, int lane, int kc) {
    const int row = rowbase + (nt << 4) + (lane & 15);
    const int r15 = row & 15;
    const int w0  = (kc << 2) + (lane >> 4);
    const char* rp = base + row * 256;
    v16bf b;
    ((uint4*)&b)[0] = *(const uint4*)(rp + (((w0)     ^ r15) << 4));
    ((uint4*)&b)[1] = *(const uint4*)(rp + (((w0 + 2) ^ r15) << 4));
    return b;
}

__device__ __forceinline__ v8f wmma_bf16(v16bf a, v16bf b, v8f c) {
    return __builtin_amdgcn_wmma_f32_16x16x32_bf16(false, a, false, b, (short)0, c, false, false);
}

// Cooperative fill of swizzled LDS weight image from a contiguous global region.
// nchunks = (#rows * 16); each chunk is 16 bytes.
__device__ __forceinline__ void fill_lds_swz(char* lds, const char* gsrc, int nchunks,
                                             int tid, int nthreads) {
    for (int c = tid; c < nchunks; c += nthreads) {
        const int row = c >> 4, w = c & 15;
        const int dst = row * 256 + ((w ^ (row & 15)) << 4);
#if USE_ASYNC_LDS
        __builtin_amdgcn_global_load_async_to_lds_b128(
            (__attribute__((address_space(1))) v4i*)(gsrc + (size_t)c * 16),
            (__attribute__((address_space(3))) v4i*)(lds + dst), 0, 0);
#else
        *(uint4*)(lds + dst) = *(const uint4*)(gsrc + (size_t)c * 16);
#endif
    }
}

__device__ __forceinline__ void lds_fill_fence() {
#if USE_ASYNC_LDS
    __builtin_amdgcn_s_wait_asynccnt(0);
#endif
    __syncthreads();
}

// ---------------- weight reformat: (H,D,DK)->(H*DK, D) etc, fp32->bf16 ----------------
__global__ void k_prep(const float* __restrict__ wq, const float* __restrict__ wk,
                       const float* __restrict__ wv,
                       __bf16* __restrict__ WqT, __bf16* __restrict__ WkT,
                       __bf16* __restrict__ WvT) {
    int i = blockIdx.x * blockDim.x + threadIdx.x;
    int stride = gridDim.x * blockDim.x;
    for (int o = i; o < 16384; o += stride) {
        int row = o >> 7, d = o & 127;
        int h = row >> 5, k = row & 31;
        WqT[o] = (__bf16)wq[(((size_t)h << 7) + d) * 32 + k];
    }
    for (int o = i; o < 81920; o += stride) {
        int r = o >> 14, rem = o & 16383;
        int row = rem >> 7, d = rem & 127;
        int h = row >> 5, k = row & 31;
        WkT[o] = (__bf16)wk[(((size_t)(h * 5 + r) << 7) + d) * 32 + k];
        WvT[o] = (__bf16)wv[(((size_t)(h * 5 + r) << 7) + d) * 32 + k];
    }
}

// ---------------- x fp32 -> bf16 ----------------
__global__ void k_xconv(const float* __restrict__ x, __bf16* __restrict__ xb, int total4) {
    int i = blockIdx.x * blockDim.x + threadIdx.x;
    int stride = gridDim.x * blockDim.x;
    for (; i < total4; i += stride) {
        float4 v = ((const float4*)x)[i];
        __bf16 t[4] = {(__bf16)v.x, (__bf16)v.y, (__bf16)v.z, (__bf16)v.w};
        ((uint2*)xb)[i] = *(const uint2*)t;
    }
}

// ---------------- pass 1: Q, K via WMMA (weights LDS-resident), logits -> Lw (N,H,5) --------
__global__ __launch_bounds__(256) void k_pass1(const __bf16* __restrict__ xb,
                                               const long long* __restrict__ nbr,
                                               const __bf16* __restrict__ WqT,  /* Wk follows */
                                               float* __restrict__ Lw, int n) {
    // rows 0..127 = WqT, rows 128..767 = WkT (5 x 128). 768 rows * 256B = 196608 B.
    __shared__ __align__(16) char lds[768 * 256];
    const int tid  = threadIdx.x;
    const int lane = tid & 31;
    const int wv   = tid >> 5;

    fill_lds_swz(lds, (const char*)WqT, 768 * 16, tid, 256);

    const int tiles = (n + 15) >> 4;
    int tile = blockIdx.x * WAVES_PER_BLOCK + wv;
    const bool active = (tile < tiles);
    if (!active) tile = tiles - 1;

    const int m16 = lane & 15;
    int node = tile * 16 + m16;
    if (node >= n) node = n - 1;

    // prefetch gathered neighbor rows while LDS fill is in flight
#pragma unroll
    for (int rr = 0; rr < 4; ++rr) {
        long long t = nbr[(size_t)node * 4 + rr];
        int g = (int)t; if (g < 0) g = 0; if (g >= n) g = n - 1;
        __builtin_prefetch(xb + (size_t)g * 128, 0, 3);
    }

    v16bf a[4];
#pragma unroll
    for (int kc = 0; kc < 4; ++kc) a[kc] = load_frag_a(xb + (size_t)node * 128, lane, kc);

    lds_fill_fence();

    // Q = Xself @ Wq^T  (16x128)
    v8f q[8];
#pragma unroll
    for (int nt = 0; nt < 8; ++nt) {
        v8f acc = vzero8();
#pragma unroll
        for (int kc = 0; kc < 4; ++kc)
            acc = wmma_bf16(a[kc], load_frag_b_lds(lds, 0, nt, lane, kc), acc);
        q[nt] = acc;
    }

    const float scale = 0.17677669529663687f;  // 1/sqrt(32)

    for (int r = 0; r < 5; ++r) {
        int grow = node;
        if (r > 0) {
            long long t = nbr[(size_t)node * 4 + (r - 1)];
            grow = (int)t; if (grow < 0) grow = 0; if (grow >= n) grow = n - 1;
        }
        v16bf ar[4];
#pragma unroll
        for (int kc = 0; kc < 4; ++kc) ar[kc] = load_frag_a(xb + (size_t)grow * 128, lane, kc);

        const int wkrow = 128 + r * 128;
        v8f kf[8];
#pragma unroll
        for (int nt = 0; nt < 8; ++nt) {
            v8f acc = vzero8();
#pragma unroll
            for (int kc = 0; kc < 4; ++kc)
                acc = wmma_bf16(ar[kc], load_frag_b_lds(lds, wkrow, nt, lane, kc), acc);
            kf[nt] = acc;
        }

        // logits: per (row m, head h) dot over 32 cols; cols live across 16-lane halves
#pragma unroll
        for (int h = 0; h < 4; ++h) {
#pragma unroll
            for (int j = 0; j < 8; ++j) {
                float s = q[2 * h][j] * kf[2 * h][j] + q[2 * h + 1][j] * kf[2 * h + 1][j];
                s += __shfl_xor(s, 1, 32);
                s += __shfl_xor(s, 2, 32);
                s += __shfl_xor(s, 4, 32);
                s += __shfl_xor(s, 8, 32);
                if (active && m16 == 0) {
                    int mrow = j + ((lane >> 4) << 3);
                    int nd = tile * 16 + mrow;
                    if (nd < n) Lw[(size_t)nd * 20 + h * 5 + r] = s * scale;
                }
            }
        }
    }
}

// ---------------- softmax over node axis: stage 1 (per-block online m,s) ----------------
__global__ __launch_bounds__(256) void k_soft_part(const float* __restrict__ Lw, int n, int npb,
                                                   float* __restrict__ pm, float* __restrict__ ps) {
    __shared__ float sm[256 * 20];
    __shared__ float ss[256 * 20];
    const int tid = threadIdx.x;
    float m[20], s[20];
#pragma unroll
    for (int c = 0; c < 20; ++c) { m[c] = -3.0e38f; s[c] = 0.0f; }

    int start = blockIdx.x * npb;
    int end = start + npb; if (end > n) end = n;
    for (int nd = start + tid; nd < end; nd += blockDim.x) {
        const float* lp = Lw + (size_t)nd * 20;
#pragma unroll
        for (int c = 0; c < 20; ++c) {
            float l = lp[c];
            if (l > m[c]) { s[c] = s[c] * __expf(m[c] - l) + 1.0f; m[c] = l; }
            else          { s[c] += __expf(l - m[c]); }
        }
    }
#pragma unroll
    for (int c = 0; c < 20; ++c) { sm[tid * 20 + c] = m[c]; ss[tid * 20 + c] = s[c]; }
    __syncthreads();
    if (tid < 20) {
        float M = -3.0e38f, S = 0.0f;
        for (int i = 0; i < 256; ++i) {
            float si = ss[i * 20 + tid];
            if (si == 0.0f) continue;
            float mi = sm[i * 20 + tid];
            if (mi > M) { S = S * __expf(M - mi) + si; M = mi; }
            else        { S += si * __expf(mi - M); }
        }
        pm[blockIdx.x * 20 + tid] = M;
        ps[blockIdx.x * 20 + tid] = S;
    }
}

// ---------------- softmax stage 2: fixed-order merge of block partials ----------------
__global__ void k_soft_final(const float* __restrict__ pm, const float* __restrict__ ps,
                             float* __restrict__ maxf, float* __restrict__ denom) {
    int c = threadIdx.x;
    if (c < 20) {
        float M = -3.0e38f, S = 0.0f;
        for (int b = 0; b < NB_SOFT; ++b) {
            float si = ps[b * 20 + c];
            if (si == 0.0f) continue;
            float mi = pm[b * 20 + c];
            if (mi > M) { S = S * __expf(M - mi) + si; M = mi; }
            else        { S += si * __expf(mi - M); }
        }
        maxf[c] = M;
        denom[c] = S;
    }
}

// ---------------- pass 3: V via WMMA (Wv LDS-resident), Z = sum_r coef * V ----------------
__global__ __launch_bounds__(256) void k_pass3(const __bf16* __restrict__ xb,
                                               const long long* __restrict__ nbr,
                                               const __bf16* __restrict__ WvT,
                                               const float* __restrict__ Lw,
                                               const float* __restrict__ maxf,
                                               const float* __restrict__ denom,
                                               float* __restrict__ out, int n) {
    __shared__ __align__(16) char lds[640 * 256];   // 5 x 128 rows of Wv^T
    const int tid  = threadIdx.x;
    const int lane = tid & 31;
    const int wv   = tid >> 5;

    fill_lds_swz(lds, (const char*)WvT, 640 * 16, tid, 256);

    const int tiles = (n + 15) >> 4;
    int tile = blockIdx.x * WAVES_PER_BLOCK + wv;
    const bool active = (tile < tiles);
    if (!active) tile = tiles - 1;

    const int m16 = lane & 15;
    int node = tile * 16 + m16;
    if (node >= n) node = n - 1;

#pragma unroll
    for (int rr = 0; rr < 4; ++rr) {
        long long t = nbr[(size_t)node * 4 + rr];
        int g = (int)t; if (g < 0) g = 0; if (g >= n) g = n - 1;
        __builtin_prefetch(xb + (size_t)g * 128, 0, 3);
    }

    float mx[20], rd[20];
#pragma unroll
    for (int c = 0; c < 20; ++c) { mx[c] = maxf[c]; rd[c] = 1.0f / denom[c]; }

    lds_fill_fence();

    v8f z[8];
#pragma unroll
    for (int nt = 0; nt < 8; ++nt) z[nt] = vzero8();

    for (int r = 0; r < 5; ++r) {
        int grow = node;
        if (r > 0) {
            long long t = nbr[(size_t)node * 4 + (r - 1)];
            grow = (int)t; if (grow < 0) grow = 0; if (grow >= n) grow = n - 1;
        }
        v16bf ar[4];
#pragma unroll
        for (int kc = 0; kc < 4; ++kc) ar[kc] = load_frag_a(xb + (size_t)grow * 128, lane, kc);

        // coefficients: exp(l - max) / denom for this lane's 8 rows x 4 heads
        float cf[4][8];
#pragma unroll
        for (int h = 0; h < 4; ++h) {
#pragma unroll
            for (int j = 0; j < 8; ++j) {
                int nd = tile * 16 + j + ((lane >> 4) << 3);
                if (nd < n) {
                    float l = Lw[(size_t)nd * 20 + h * 5 + r];
                    cf[h][j] = __expf(l - mx[h * 5 + r]) * rd[h * 5 + r];
                } else {
                    cf[h][j] = 0.0f;
                }
            }
        }

#pragma unroll
        for (int nt = 0; nt < 8; ++nt) {
            v8f vac = vzero8();
#pragma unroll
            for (int kc = 0; kc < 4; ++kc)
                vac = wmma_bf16(ar[kc], load_frag_b_lds(lds, r * 128, nt, lane, kc), vac);
            const int h = nt >> 1;
#pragma unroll
            for (int j = 0; j < 8; ++j)
                z[nt][j] = fmaf(cf[h][j], vac[j], z[nt][j]);
        }
    }

#pragma unroll
    for (int nt = 0; nt < 8; ++nt) {
#pragma unroll
        for (int j = 0; j < 8; ++j) {
            int nd = tile * 16 + j + ((lane >> 4) << 3);
            if (active && nd < n)
                out[(size_t)nd * 128 + nt * 16 + m16] = z[nt][j];
        }
    }
}

extern "C" void kernel_launch(void* const* d_in, const int* in_sizes, int n_in,
                              void* d_out, int out_size, void* d_ws, size_t ws_size,
                              hipStream_t stream) {
    const float*     x   = (const float*)d_in[0];
    const long long* nbr = (const long long*)d_in[1];
    const float*     wq  = (const float*)d_in[2];
    const float*     wk  = (const float*)d_in[3];
    const float*     wv  = (const float*)d_in[4];
    float* out = (float*)d_out;
    const int n = in_sizes[0] / 128;

    char* w = (char*)d_ws;
    size_t off = 0;
    __bf16* xb = (__bf16*)(w + off);  off += (size_t)n * 128 * 2; off = (off + 255) & ~(size_t)255;
    __bf16* WqT = (__bf16*)(w + off); off += 32768;     // Wk must stay contiguous after Wq
    __bf16* WkT = (__bf16*)(w + off); off += 163840;
    __bf16* WvT = (__bf16*)(w + off); off += 163840;
    float*  Lw  = (float*)(w + off);  off += (size_t)n * 20 * 4; off = (off + 255) & ~(size_t)255;
    float*  pm  = (float*)(w + off);  off += (size_t)NB_SOFT * 20 * 4;
    float*  ps  = (float*)(w + off);  off += (size_t)NB_SOFT * 20 * 4;
    float*  maxf  = (float*)(w + off); off += 256;
    float*  denom = (float*)(w + off); off += 256;
    (void)ws_size; (void)n_in; (void)out_size;

    k_prep<<<96, 256, 0, stream>>>(wq, wk, wv, WqT, WkT, WvT);

    int total4 = (n * 128) / 4;
    int gx = (total4 + 255) / 256; if (gx > 4096) gx = 4096;
    k_xconv<<<gx, 256, 0, stream>>>(x, xb, total4);

    int tiles = (n + 15) / 16;
    int blocks = (tiles + WAVES_PER_BLOCK - 1) / WAVES_PER_BLOCK;
    k_pass1<<<blocks, 256, 0, stream>>>(xb, nbr, WqT, Lw, n);

    int npb = (n + NB_SOFT - 1) / NB_SOFT;
    k_soft_part<<<NB_SOFT, 256, 0, stream>>>(Lw, n, npb, pm, ps);
    k_soft_final<<<1, 32, 0, stream>>>(pm, ps, maxf, denom);

    k_pass3<<<blocks, 256, 0, stream>>>(xb, nbr, WvT, Lw, maxf, denom, out, n);
}
